// SelfAttention_65300682768451
// MI455X (gfx1250) — matrix operand order repaired and checked by reading, not compiled, not run on hardware
//
#include <hip/hip_runtime.h>

#define S_LEN 2048
#define H_DIM 1024
#define NHEAD 16
#define HDIM  64
#define BATCH 2

typedef __attribute__((ext_vector_type(16))) __bf16       v16bf;
typedef __attribute__((ext_vector_type(16))) _Float16     v16h;
typedef __attribute__((ext_vector_type(8)))  float        v8f;
typedef __attribute__((ext_vector_type(4)))  unsigned int u32x4;
typedef __attribute__((ext_vector_type(4)))  float        f32x4;

union Frag  { v16bf v; u32x4 u[2]; };
union FragH { v16h  v; u32x4 u[2]; };

__device__ inline unsigned short f2bf(float x) {
    unsigned int u = __float_as_uint(x);
    u += 0x7FFFu + ((u >> 16) & 1u);   // round-to-nearest-even
    return (unsigned short)(u >> 16);
}

__device__ inline v8f wmma_bf16(const Frag& a, const Frag& b, v8f c) {
    return __builtin_amdgcn_wmma_f32_16x16x32_bf16(
        false, a.v, false, b.v, (short)0, c, false, false);
}
__device__ inline v8f wmma_f16(const FragH& a, const FragH& b, v8f c) {
    return __builtin_amdgcn_wmma_f32_16x16x32_f16(
        false, a.v, false, b.v, (short)0, c, false, false);
}

// ---------------------------------------------------------------------------
// Kernel 0: one-shot f32 -> bf16 conversion (bandwidth bound).
// Removes all conversion VALU from the GEMM inner loop and the 16-64x
// redundant re-conversion of X / W across output tiles.
// ---------------------------------------------------------------------------
__global__ __launch_bounds__(256) void cvt_bf16_kernel(
    const float* __restrict__ src, unsigned short* __restrict__ dst, int n8)
{
    const int i = blockIdx.x * 256 + threadIdx.x;   // 8 elements per thread
    if (i >= n8) return;
    const f32x4* p = (const f32x4*)(src + (size_t)i * 8);
    f32x4 a = p[0], b = p[1];
    union { unsigned short h[8]; u32x4 u; } o;
    #pragma unroll
    for (int e = 0; e < 4; ++e) { o.h[e] = f2bf(a[e]); o.h[4 + e] = f2bf(b[e]); }
    *(u32x4*)(dst + (size_t)i * 8) = o.u;
}

// ---------------------------------------------------------------------------
// Kernel 1: QKV projection on pre-converted bf16. out = X @ W.T + b.
// 64x64 tile / block (4 waves), K-stage 64 staged to LDS with CDNA5 async
// global->LDS copies (no VGPR round trip), 8 WMMAs per wave per stage.
// Q scaled by 1/sqrt(HD) -> bf16 [B,NH,S,HD]; K -> bf16 [B,NH,S,HD];
// V -> f16 transposed [B,NH,HD,S] so the P@V B-frag is contiguous.
// ---------------------------------------------------------------------------
__global__ __launch_bounds__(128) void qkv_gemm_kernel(
    const unsigned short* __restrict__ Xb,
    const unsigned short* __restrict__ Wqb,
    const unsigned short* __restrict__ Wkb,
    const unsigned short* __restrict__ Wvb,
    const float* __restrict__ bq, const float* __restrict__ bk,
    const float* __restrict__ bv,
    unsigned short* __restrict__ Qb, unsigned short* __restrict__ Kb,
    _Float16* __restrict__ Vtb)
{
    __shared__ __align__(16) unsigned short xs[64 * 64];
    __shared__ __align__(16) unsigned short wt[64 * 64];

    const int proj = blockIdx.z;
    const unsigned short* __restrict__ W =
        (proj == 0) ? Wqb : (proj == 1) ? Wkb : Wvb;
    const float* __restrict__ bias = (proj == 0) ? bq : (proj == 1) ? bk : bv;

    const int m0   = blockIdx.y * 64;
    const int n0   = blockIdx.x * 64;
    const int tid  = threadIdx.x;
    const int wave = tid >> 5;
    const int lane = tid & 31;
    const int half = lane >> 4;
    const int lm   = lane & 15;

    // raw LDS byte offsets for the async-copy destination operand
    const unsigned xs_base = (unsigned)(unsigned long long)&xs[0];
    const unsigned wt_base = (unsigned)(unsigned long long)&wt[0];

    v8f acc[4];
    for (int c = 0; c < 4; ++c)
        for (int r = 0; r < 8; ++r) acc[c][r] = 0.f;

    for (int k0 = 0; k0 < H_DIM; k0 += 64) {
        // stage 64x64 bf16 X-tile and W-tile: 512 16B chunks each,
        // 4 chunks/thread/array, issued as async global->LDS b128 copies
        #pragma unroll
        for (int i = 0; i < 4; ++i) {
            const int cid = i * 128 + tid;        // 0..511
            const int row = cid >> 3;             // 0..63
            const int col = (cid & 7) * 8;        // halves (16B chunks)
            const unsigned lofs = (unsigned)(row * 64 + col) * 2u;
            const unsigned lx = xs_base + lofs;
            const unsigned lw = wt_base + lofs;
            const unsigned short* gx = &Xb[(size_t)(m0 + row) * H_DIM + k0 + col];
            const unsigned short* gw = &W [(size_t)(n0 + row) * H_DIM + k0 + col];
            asm volatile("global_load_async_to_lds_b128 %0, %1, off"
                         :: "v"(lx), "v"(gx) : "memory");
            asm volatile("global_load_async_to_lds_b128 %0, %1, off"
                         :: "v"(lw), "v"(gw) : "memory");
        }
        asm volatile("s_wait_asynccnt 0x0" ::: "memory");
        __syncthreads();

        #pragma unroll
        for (int kk = 0; kk < 2; ++kk) {
            Frag a;   // A-frag: row = wave*16+lm, K = kk*32 + half*16..+15
            a.u[0] = *(const u32x4*)&xs[(wave * 16 + lm) * 64 + kk * 32 + half * 16];
            a.u[1] = *(const u32x4*)&xs[(wave * 16 + lm) * 64 + kk * 32 + half * 16 + 8];
            #pragma unroll
            for (int c = 0; c < 4; ++c) {
                Frag b;   // B-frag: col = c*16+lm, K = kk*32 + half*16..+15
                const u32x4* pb = (const u32x4*)&wt[(c * 16 + lm) * 64 + kk * 32 + half * 16];
                b.u[0] = pb[0]; b.u[1] = pb[1];
                acc[c] = wmma_bf16(a, b, acc[c]);
            }
        }
        __syncthreads();
    }

    #pragma unroll
    for (int c = 0; c < 4; ++c) {
        const int   n  = n0 + c * 16 + lm;
        const float bn = bias[n];
        const int   h  = n >> 6, d = n & 63;
        #pragma unroll
        for (int r = 0; r < 8; ++r) {
            const int m  = m0 + wave * 16 + half * 8 + r;   // C-layout row
            const int bb = m >> 11;
            const int s  = m & (S_LEN - 1);
            const float  val = acc[c][r] + bn;
            const size_t hb  = (size_t)(bb * NHEAD + h);
            if (proj == 0)      Qb [(hb * S_LEN + s) * HDIM + d] = f2bf(val * 0.125f);
            else if (proj == 1) Kb [(hb * S_LEN + s) * HDIM + d] = f2bf(val);
            else                Vtb[(hb * HDIM + d) * S_LEN + s] = (_Float16)val;
        }
    }
}

// ---------------------------------------------------------------------------
// Kernel 2: flash attention. 128 threads (4 waves) = 64 q-rows per
// (batch, head). Q.K^T in bf16 WMMA, online softmax in C-fragment layout,
// probs -> f16 via per-wave LDS, P@V in f16 WMMA, f32 accumulation.
// ---------------------------------------------------------------------------
__global__ __launch_bounds__(128) void attention_kernel(
    const unsigned short* __restrict__ Qb,
    const unsigned short* __restrict__ Kb,
    const _Float16* __restrict__ Vtb,
    const float* __restrict__ mask,
    float* __restrict__ out)
{
    __shared__ __align__(16) _Float16 plds[4 * 16 * 64];  // per-wave P tile

    const int qt   = blockIdx.x;
    const int h    = blockIdx.y;
    const int b    = blockIdx.z;
    const int tid  = threadIdx.x;
    const int wave = tid >> 5;
    const int lane = tid & 31;
    const int half = lane >> 4;
    const int lm   = lane & 15;

    const size_t hb = (size_t)(b * NHEAD + h);
    const unsigned short* Qh = Qb  + hb * S_LEN * HDIM;
    const unsigned short* Kh = Kb  + hb * S_LEN * HDIM;
    const _Float16*       Vh = Vtb + hb * HDIM * S_LEN;   // [d][s]
    const float*          mk = mask + (size_t)b * S_LEN;

    const int qrow = qt * 64 + wave * 16 + lm;
    Frag qa[2];
    #pragma unroll
    for (int kk = 0; kk < 2; ++kk) {
        const unsigned short* p = &Qh[(size_t)qrow * HDIM + kk * 32];
        qa[kk].u[0] = *(const u32x4*)&p[half * 16];
        qa[kk].u[1] = *(const u32x4*)&p[half * 16 + 8];
    }

    float mrun[8], lrun[8];
    v8f o[4];
    #pragma unroll
    for (int r = 0; r < 8; ++r) { mrun[r] = -1e30f; lrun[r] = 0.f; }
    for (int c = 0; c < 4; ++c)
        for (int r = 0; r < 8; ++r) o[c][r] = 0.f;

    _Float16* pl = &plds[wave * 16 * 64];

    for (int j0 = 0; j0 < S_LEN; j0 += 64) {
        v8f sc[4];
        for (int c = 0; c < 4; ++c)
            for (int r = 0; r < 8; ++r) sc[c][r] = 0.f;
        #pragma unroll
        for (int kk = 0; kk < 2; ++kk) {
            #pragma unroll
            for (int c = 0; c < 4; ++c) {
                Frag kb;
                const u32x4* pb = (const u32x4*)&Kh[(size_t)(j0 + c * 16 + lm) * HDIM + kk * 32 + half * 16];
                kb.u[0] = pb[0]; kb.u[1] = pb[1];
                sc[c] = wmma_bf16(qa[kk], kb, sc[c]);
            }
        }
        #pragma unroll
        for (int c = 0; c < 4; ++c) {
            const float mc = mk[j0 + c * 16 + lm];
            #pragma unroll
            for (int r = 0; r < 8; ++r) sc[c][r] += mc;
        }
        #pragma unroll
        for (int r = 0; r < 8; ++r) {
            float v = mrun[r];
            #pragma unroll
            for (int c = 0; c < 4; ++c) v = fmaxf(v, sc[c][r]);
            for (int off = 8; off > 0; off >>= 1) v = fmaxf(v, __shfl_xor(v, off, 32));
            float ps = 0.f;
            #pragma unroll
            for (int c = 0; c < 4; ++c) {
                float p = __expf(sc[c][r] - v);
                sc[c][r] = p; ps += p;
            }
            for (int off = 8; off > 0; off >>= 1) ps += __shfl_xor(ps, off, 32);
            const float resc = __expf(mrun[r] - v);
            lrun[r] = lrun[r] * resc + ps;
            mrun[r] = v;
            #pragma unroll
            for (int c = 0; c < 4; ++c) o[c][r] *= resc;
        }
        // C-layout -> A-layout via per-wave LDS, probs in f16 (1 cvt/elem)
        #pragma unroll
        for (int c = 0; c < 4; ++c)
            #pragma unroll
            for (int r = 0; r < 8; ++r)
                pl[(half * 8 + r) * 64 + c * 16 + lm] = (_Float16)sc[c][r];
        asm volatile("s_wait_dscnt 0" ::: "memory");
        #pragma unroll
        for (int kk = 0; kk < 2; ++kk) {
            FragH pa;
            pa.u[0] = *(const u32x4*)&pl[lm * 64 + kk * 32 + half * 16];
            pa.u[1] = *(const u32x4*)&pl[lm * 64 + kk * 32 + half * 16 + 8];
            #pragma unroll
            for (int c = 0; c < 4; ++c) {
                FragH vb;
                const u32x4* pv = (const u32x4*)&Vh[(size_t)(c * 16 + lm) * S_LEN + j0 + kk * 32 + half * 16];
                vb.u[0] = pv[0]; vb.u[1] = pv[1];
                o[c] = wmma_f16(pa, vb, o[c]);
            }
        }
    }

    #pragma unroll
    for (int c = 0; c < 4; ++c) {
        const int d = c * 16 + lm;
        #pragma unroll
        for (int r = 0; r < 8; ++r) {
            const int s = qt * 64 + wave * 16 + half * 8 + r;
            out[((size_t)b * S_LEN + s) * H_DIM + h * HDIM + d] = o[c][r] / lrun[r];
        }
    }
}

extern "C" void kernel_launch(void* const* d_in, const int* in_sizes, int n_in,
                              void* d_out, int out_size, void* d_ws, size_t ws_size,
                              hipStream_t stream) {
    const float* X    = (const float*)d_in[0];
    const float* mask = (const float*)d_in[1];
    const float* Wq   = (const float*)d_in[2];
    const float* bq   = (const float*)d_in[3];
    const float* Wk   = (const float*)d_in[4];
    const float* bk   = (const float*)d_in[5];
    const float* Wv   = (const float*)d_in[6];
    const float* bv   = (const float*)d_in[7];
    float* out = (float*)d_out;

    const size_t xel = (size_t)BATCH * S_LEN * H_DIM;   // 4M
    const size_t wel = (size_t)H_DIM * H_DIM;           // 1M
    unsigned short* Xb  = (unsigned short*)d_ws;
    unsigned short* Wqb = Xb + xel;
    unsigned short* Wkb = Wqb + wel;
    unsigned short* Wvb = Wkb + wel;
    unsigned short* Qb  = Wvb + wel;
    unsigned short* Kb  = Qb + xel;
    _Float16*       Vtb = (_Float16*)(Kb + xel);

    cvt_bf16_kernel<<<dim3((unsigned)(xel / 8 / 256)), dim3(256), 0, stream>>>(X,  Xb,  (int)(xel / 8));
    cvt_bf16_kernel<<<dim3((unsigned)(wel / 8 / 256)), dim3(256), 0, stream>>>(Wq, Wqb, (int)(wel / 8));
    cvt_bf16_kernel<<<dim3((unsigned)(wel / 8 / 256)), dim3(256), 0, stream>>>(Wk, Wkb, (int)(wel / 8));
    cvt_bf16_kernel<<<dim3((unsigned)(wel / 8 / 256)), dim3(256), 0, stream>>>(Wv, Wvb, (int)(wel / 8));

    dim3 g1(H_DIM / 64, (BATCH * S_LEN) / 64, 3);
    qkv_gemm_kernel<<<g1, dim3(128), 0, stream>>>(Xb, Wqb, Wkb, Wvb, bq, bk, bv, Qb, Kb, Vtb);

    dim3 g2(S_LEN / 64, NHEAD, BATCH);
    attention_kernel<<<g2, dim3(128), 0, stream>>>(Qb, Kb, Vtb, mask, out);
}
